// StackT5Attention_12421045420109
// MI455X (gfx1250) — matrix operand order repaired and checked by reading, not compiled, not run on hardware
//
#include <hip/hip_runtime.h>
#include <cstdint>
#include <cstddef>

// ---------------------------------------------------------------------------
// Problem constants (from reference)
// ---------------------------------------------------------------------------
#define BB   2
#define SS   2048
#define HID  2048
#define HH   16
#define DD   128
// net score scale: (1/LAYER_IDX/sqrt(D)) * LAYER_IDX = 1/sqrt(128)
#define SOFTMAX_SCALE 0.08838834764831845f
#define MASK_VALUE    (-10000.0f)

typedef __attribute__((ext_vector_type(16))) __bf16 v16bf;
typedef __attribute__((ext_vector_type(8)))  float  v8f;

__device__ __forceinline__ v8f wmma_bf16(v16bf a, v16bf b, v8f c) {
  // v_wmma_f32_16x16x32_bf16: (neg_a, A, neg_b, B, c_mod, C, reuse_a, reuse_b)
  return __builtin_amdgcn_wmma_f32_16x16x32_bf16(false, a, false, b, (short)0, c,
                                                 false, false);
}

// Assemble a 16-element bf16 fragment from two contiguous 16-byte runs.
__device__ __forceinline__ v16bf ld_frag(const __bf16* p0, const __bf16* p1) {
  union { int4 q[2]; v16bf v; } u;
  u.q[0] = *(const int4*)p0;
  u.q[1] = *(const int4*)p1;
  return u.v;
}

// CDNA5 async global->LDS copy, 16 bytes per lane (ASYNCcnt-tracked, no VGPR
// round trip). LDS destination address = low 32 bits of the generic pointer.
__device__ __forceinline__ void async_copy_b128(void* lds_dst, const void* gsrc) {
  uint32_t lds = (uint32_t)(uintptr_t)lds_dst;
  asm volatile("global_load_async_to_lds_b128 %0, %1, off"
               :: "v"(lds), "v"(gsrc) : "memory");
}
__device__ __forceinline__ void wait_async0() {
  asm volatile("s_wait_asynccnt 0x0" ::: "memory");
}

// ---------------------------------------------------------------------------
// Kernel 1: fp32 -> bf16 conversion (vectorized)
// ---------------------------------------------------------------------------
__global__ void cvt_f32_to_bf16(const float* __restrict__ in,
                                __bf16* __restrict__ out, int n) {
  int i = (blockIdx.x * blockDim.x + threadIdx.x) * 4;
  if (i + 3 < n) {
    float4 f = *(const float4*)(in + i);
    union { __bf16 h[4]; uint2 u; } o;
    o.h[0] = (__bf16)f.x; o.h[1] = (__bf16)f.y;
    o.h[2] = (__bf16)f.z; o.h[3] = (__bf16)f.w;
    *(uint2*)(out + i) = o.u;
  } else {
    for (; i < n; ++i) out[i] = (__bf16)in[i];
  }
}

// ---------------------------------------------------------------------------
// Kernel 2: bf16 WMMA GEMM, C = A(MxK) * B(KxN) + bias, row-major.
// Block tile 128x64, 256 threads = 8 waves laid out 4(M) x 2(N);
// each wave computes a 32x32 patch = 2x2 v_wmma_f32_16x16x32_bf16 tiles.
// Double-buffered LDS: A staged with global_load_async_to_lds_b128,
// B transposed into LDS through VGPRs; prefetch tile t+1 under tile t's WMMAs.
// ---------------------------------------------------------------------------
template <typename OutT>
__global__ __launch_bounds__(256, 1)
void gemm_bf16_wmma(const __bf16* __restrict__ A,
                    const __bf16* __restrict__ B,
                    const float* __restrict__ bias,
                    OutT* __restrict__ C,
                    int M, int N, int K) {
  __shared__ alignas(16) __bf16 As[2][128][32];   // [buf][row][k]   16 KB
  __shared__ alignas(16) __bf16 Bst[2][64][32];   // [buf][col][k]    8 KB

  const int tid  = threadIdx.x;
  const int wave = tid >> 5;
  const int lane = tid & 31;
  const int l15  = lane & 15;
  const int lh   = lane >> 4;
  const int wm   = wave >> 1;       // 0..3
  const int wn   = wave & 1;        // 0..1
  const int row0 = blockIdx.y * 128;
  const int col0 = blockIdx.x * 64;

  v8f acc[2][2] = {};

  auto stage = [&](int k0, int buf) {
    // A tile: 128 rows x 32 k = 512 x 16B chunks, 2 per thread, async to LDS
#pragma unroll
    for (int p = 0; p < 2; ++p) {
      int idx = tid + p * 256;             // 0..511
      int r   = idx >> 2;                  // 0..127
      int c   = (idx & 3) * 8;             // 0,8,16,24
      async_copy_b128(&As[buf][r][c],
                      A + (size_t)(row0 + r) * K + k0 + c);
    }
    // B tile transposed: B[k0+kk][col0+n] -> Bst[n][kk]
    int kk = tid >> 3;                     // 0..31
    int n0 = (tid & 7) * 8;                // 0..56
    const __bf16* src = B + (size_t)(k0 + kk) * N + col0 + n0;
#pragma unroll
    for (int i = 0; i < 8; ++i) Bst[buf][n0 + i][kk] = src[i];
  };

  const int ntiles = K >> 5;
  stage(0, 0);  // prefetch tile 0

  for (int t = 0; t < ntiles; ++t) {
    const int cb = t & 1;
    wait_async0();          // this wave's async copies into buf cb done
    __syncthreads();        // all waves' staging (async + ds) visible
    if (t + 1 < ntiles) stage((t + 1) << 5, cb ^ 1);

    // ---- fragments + WMMA on buffer cb
#pragma unroll
    for (int mt = 0; mt < 2; ++mt) {
      // A frag: row = wm*32+mt*16+l15; k: [lh*8 .. +7] and [16+lh*8 .. +7]
      const __bf16* ar = &As[cb][wm * 32 + mt * 16 + l15][0];
      v16bf a = ld_frag(ar + lh * 8, ar + 16 + lh * 8);
#pragma unroll
      for (int nt = 0; nt < 2; ++nt) {
        // B frag: col = wn*32+nt*16+l15; k: [lh*8 .. +7] and [16+lh*8 .. +7]
        const __bf16* br = &Bst[cb][wn * 32 + nt * 16 + l15][0];
        v16bf b = ld_frag(br + lh * 8, br + 16 + lh * 8);
        acc[mt][nt] = wmma_bf16(a, b, acc[mt][nt]);
      }
    }
  }

  // ---- epilogue: C layout: lane holds N=l15(+tile), rows M = lh*8 + r
#pragma unroll
  for (int mt = 0; mt < 2; ++mt)
#pragma unroll
    for (int nt = 0; nt < 2; ++nt) {
      int col = col0 + wn * 32 + nt * 16 + l15;
      float bv = bias[col];
#pragma unroll
      for (int r = 0; r < 8; ++r) {
        int row = row0 + wm * 32 + mt * 16 + lh * 8 + r;
        C[(size_t)row * N + col] = (OutT)(acc[mt][nt][r] + bv);
      }
    }
}

// ---------------------------------------------------------------------------
// Kernel 3: fused flash attention (multi-query: one shared K/V head).
// grid = (S/128, H, B); 256 threads = 8 waves; wave w owns query rows
// [qt*128 + w*16, +16). Q fragments live in registers for the whole kernel.
// Double-buffered 32-key chunks: K staged async (row-major), V transposed via
// VGPRs; QK^T (8 WMMAs) -> online softmax (shfl in 16-lane halves) ->
// P re-layout through per-wave LDS patch -> PV (8 WMMAs).
// ---------------------------------------------------------------------------
__global__ __launch_bounds__(256, 1)
void flash_attn_bf16(const __bf16* __restrict__ q,
                     const __bf16* __restrict__ kv,
                     __bf16* __restrict__ attn) {
  __shared__ alignas(16) __bf16 Ks[2][32][DD];     // [buf][key][d]  16 KB
  __shared__ alignas(16) __bf16 Vst[2][DD][32];    // [buf][d][key]  16 KB
  __shared__ alignas(16) __bf16 Ps[8][16][32];     // per-wave P      8 KB

  const int tid  = threadIdx.x;
  const int wave = tid >> 5;
  const int lane = tid & 31;
  const int l15  = lane & 15;
  const int lh   = lane >> 4;

  const int qt = blockIdx.x;    // query tile (128 rows)
  const int h  = blockIdx.y;    // head
  const int b  = blockIdx.z;    // batch

  // ---- load Q fragments (16 rows x 128 d -> 4 A-fragments of K=32)
  const size_t qrow_stride = (size_t)HH * DD;
  const __bf16* qrowp = q + ((size_t)b * SS + qt * 128 + wave * 16 + l15) * qrow_stride
                          + (size_t)h * DD;
  v16bf qf[4];
#pragma unroll
  for (int c = 0; c < 4; ++c)
    qf[c] = ld_frag(qrowp + c * 32 + lh * 8, qrowp + c * 32 + 16 + lh * 8);

  v8f accO[8] = {};
  float m[8], lsum[8];
#pragma unroll
  for (int r = 0; r < 8; ++r) { m[r] = -1e30f; lsum[r] = 0.0f; }

  const __bf16* kvb = kv + (size_t)b * SS * (2 * DD);
  const int nchunks = (qt + 1) * 4;  // keys processed: (qt+1)*128 (causal)

  auto stage_kv = [&](int j, int buf) {
#pragma unroll
    for (int p = 0; p < 2; ++p) {
      int idx = tid + p * 256;           // 0..511
      int r   = idx >> 4;                // key row 0..31
      int c   = (idx & 15) * 8;          // d 0..120
      const __bf16* krow = kvb + (size_t)(j * 32 + r) * (2 * DD);
      async_copy_b128(&Ks[buf][r][c], krow + c);        // K: async, row-major
      alignas(16) __bf16 tmp[8];                        // V: transpose via VGPR
      *(int4*)tmp = *(const int4*)(krow + DD + c);
#pragma unroll
      for (int i = 0; i < 8; ++i) Vst[buf][c + i][r] = tmp[i];
    }
  };

  stage_kv(0, 0);  // prefetch chunk 0

  for (int j = 0; j < nchunks; ++j) {
    const int cb = j & 1;
    wait_async0();
    __syncthreads();
    if (j + 1 < nchunks) stage_kv(j + 1, cb ^ 1);

    // ---- scores: two 16-key subtiles, contraction over d=128 (4 WMMAs each)
    v8f sc[2] = {};
#pragma unroll
    for (int st = 0; st < 2; ++st)
#pragma unroll
      for (int c = 0; c < 4; ++c) {
        const __bf16* br = &Ks[cb][st * 16 + l15][c * 32];
        sc[st] = wmma_bf16(qf[c], ld_frag(br + lh * 8, br + 16 + lh * 8), sc[st]);
      }

    // ---- scale + causal mask (C layout: N=key=l15, M=lh*8+r)
#pragma unroll
    for (int st = 0; st < 2; ++st) {
      int key = j * 32 + st * 16 + l15;
#pragma unroll
      for (int r = 0; r < 8; ++r) {
        int qrow = qt * 128 + wave * 16 + lh * 8 + r;
        float s  = sc[st][r] * SOFTMAX_SCALE;
        sc[st][r] = (key <= qrow) ? s : MASK_VALUE;
      }
    }

    // ---- online softmax per row (reduce across the 16 lanes of each half)
#pragma unroll
    for (int r = 0; r < 8; ++r) {
      float tm = fmaxf(sc[0][r], sc[1][r]);
#pragma unroll
      for (int msk = 1; msk < 16; msk <<= 1)
        tm = fmaxf(tm, __shfl_xor(tm, msk, 32));
      float mnew  = fmaxf(m[r], tm);
      float alpha = __expf(m[r] - mnew);
      m[r] = mnew;
      float p0 = __expf(sc[0][r] - mnew);
      float p1 = __expf(sc[1][r] - mnew);
      sc[0][r] = p0; sc[1][r] = p1;
      float ps = p0 + p1;
#pragma unroll
      for (int msk = 1; msk < 16; msk <<= 1)
        ps += __shfl_xor(ps, msk, 32);
      lsum[r] = lsum[r] * alpha + ps;
#pragma unroll
      for (int dt = 0; dt < 8; ++dt) accO[dt][r] *= alpha;
    }

    // ---- re-layout P: C-fragment -> LDS -> A-fragment (per-wave patch)
#pragma unroll
    for (int st = 0; st < 2; ++st)
#pragma unroll
      for (int r = 0; r < 8; ++r)
        Ps[wave][lh * 8 + r][st * 16 + l15] = (__bf16)sc[st][r];
    const __bf16* pr = &Ps[wave][l15][0];
    v16bf pa = ld_frag(pr + lh * 8, pr + 16 + lh * 8);  // same-wave DS order

    // ---- PV: O(16x128) += P(16x32) @ V(32x128), 8 WMMAs
#pragma unroll
    for (int dt = 0; dt < 8; ++dt) {
      const __bf16* vr = &Vst[cb][dt * 16 + l15][0];
      accO[dt] = wmma_bf16(pa, ld_frag(vr + lh * 8, vr + 16 + lh * 8), accO[dt]);
    }
  }

  // ---- epilogue: normalize and store bf16 attn (token-major, [token][h*D+d])
  const size_t obase = ((size_t)b * SS + qt * 128 + wave * 16) * qrow_stride
                       + (size_t)h * DD;
#pragma unroll
  for (int dt = 0; dt < 8; ++dt)
#pragma unroll
    for (int r = 0; r < 8; ++r) {
      int row = lh * 8 + r;
      attn[obase + (size_t)row * qrow_stride + dt * 16 + l15] =
          (__bf16)(accO[dt][r] / lsum[r]);
    }
}

// ---------------------------------------------------------------------------
// Host-side orchestration
// ---------------------------------------------------------------------------
extern "C" void kernel_launch(void* const* d_in, const int* in_sizes, int n_in,
                              void* d_out, int out_size, void* d_ws, size_t ws_size,
                              hipStream_t stream) {
  (void)in_sizes; (void)n_in; (void)out_size; (void)ws_size;
  const float* hidden = (const float*)d_in[0];
  // d_in[1] = attention_mask (causal, known statically -> unused)
  const float* Wq  = (const float*)d_in[2];
  const float* bq  = (const float*)d_in[3];
  const float* Wkv = (const float*)d_in[4];
  const float* bkv = (const float*)d_in[5];
  const float* Wp  = (const float*)d_in[6];
  const float* bp  = (const float*)d_in[7];
  float* out = (float*)d_out;

  // workspace carve-up (bf16 staging buffers, ~67 MB total)
  char* ws = (char*)d_ws;
  auto take = [&](size_t elems) { __bf16* p = (__bf16*)ws; ws += elems * sizeof(__bf16); return p; };
  __bf16* h_bf    = take((size_t)BB * SS * HID);       // hidden
  __bf16* wq_bf   = take((size_t)HID * HH * DD);       // Wq
  __bf16* wkv_bf  = take((size_t)HID * 2 * DD);        // Wkv
  __bf16* wp_bf   = take((size_t)HH * DD * HID);       // Wp
  __bf16* q_bf    = take((size_t)BB * SS * HH * DD);   // q
  __bf16* kv_bf   = take((size_t)BB * SS * 2 * DD);    // kv (k|v interleaved)
  __bf16* attn_bf = take((size_t)BB * SS * HH * DD);   // attention output

  auto cvt = [&](const float* src, __bf16* dst, size_t n) {
    int blocks = (int)((n / 4 + 255) / 256);
    cvt_f32_to_bf16<<<blocks, 256, 0, stream>>>(src, dst, (int)n);
  };
  cvt(hidden, h_bf,  (size_t)BB * SS * HID);
  cvt(Wq,     wq_bf, (size_t)HID * HH * DD);
  cvt(Wkv,    wkv_bf,(size_t)HID * 2 * DD);
  cvt(Wp,     wp_bf, (size_t)HH * DD * HID);

  const int M = BB * SS;  // 4096 tokens

  // Q projection: (4096 x 2048) @ (2048 x 2048) + bq -> bf16
  gemm_bf16_wmma<__bf16><<<dim3((HH * DD) / 64, M / 128), 256, 0, stream>>>(
      h_bf, wq_bf, bq, q_bf, M, HH * DD, HID);

  // KV projection: (4096 x 2048) @ (2048 x 256) + bkv -> bf16
  gemm_bf16_wmma<__bf16><<<dim3((2 * DD) / 64, M / 128), 256, 0, stream>>>(
      h_bf, wkv_bf, bkv, kv_bf, M, 2 * DD, HID);

  // Fused flash attention
  flash_attn_bf16<<<dim3(SS / 128, HH, BB), 256, 0, stream>>>(q_bf, kv_bf, attn_bf);

  // Output projection: (4096 x 2048) @ (2048 x 2048) + bp -> fp32 out
  gemm_bf16_wmma<float><<<dim3(HID / 64, M / 128), 256, 0, stream>>>(
      attn_bf, wp_bf, bp, out, M, HID, HH * DD);
}
